// MHA_18442589569914
// MI455X (gfx1250) — matrix operand order repaired and checked by reading, not compiled, not run on hardware
//
#include <hip/hip_runtime.h>
#include <hip/hip_bf16.h>
#include <math.h>

#define B_   2
#define S_   2048
#define D_   1024
#define H_   16
#define DH_  64
#define BS_  (B_ * S_)                           /* 4096 rows */
#define PROJ_ELEMS ((size_t)B_ * H_ * S_ * DH_)  /* 4,194,304 */
#define WQKV_ELEMS ((size_t)H_ * D_ * DH_)       /* 1,048,576 */
#define X_ELEMS    ((size_t)BS_ * D_)            /* 4,194,304 */

typedef __bf16 bf16_t;
typedef __attribute__((ext_vector_type(16))) __bf16 v16bf;
typedef __attribute__((ext_vector_type(8)))  __bf16 bf16x8;
typedef __attribute__((ext_vector_type(8)))  float  v8f;
typedef __attribute__((ext_vector_type(8)))  float  f32x8;

static __device__ __forceinline__ v8f wmma_bf16(v16bf a, v16bf b, v8f c) {
  // D = A(16x32) * B(32x16) + C, f32 accumulate
  return __builtin_amdgcn_wmma_f32_16x16x32_bf16(false, a, false, b,
                                                 (short)0, c, false, false);
}

// ---- Fragment loaders (contiguous k order, same for A and B) ------------
// A 16x32 bf16: lane L holds row m=L&15; K-slice [kg,kg+15],
// kg = (L>>4)*16. Element e maps to k = kg+e.
static __device__ __forceinline__ v16bf
load_Afrag_bf16(const bf16_t* __restrict__ A, int lda, int m0, int k0, int lane) {
  const int m  = m0 + (lane & 15);
  const int kg = (lane >> 4) * 16;
  const bf16_t* row = A + (size_t)m * lda + k0 + kg;
  bf16x8 lo = *(const bf16x8*)(row);
  bf16x8 hi = *(const bf16x8*)(row + 8);
  v16bf a;
#pragma unroll
  for (int i = 0; i < 8; ++i) { a[i] = lo[i]; a[8 + i] = hi[i]; }
  return a;
}

// B 32x16 bf16 where B[k][n] = src[n0+n][k0+k] (src row-major, contiguous k).
// Lane L: n = L&15, K-slice [kg, kg+15], kg = (L>>4)*16 -> one 32B vector load.
static __device__ __forceinline__ v16bf
load_Bfrag_rows_bf16(const bf16_t* __restrict__ src, int ld, int n0, int k0, int lane) {
  const int n  = n0 + (lane & 15);
  const int kg = (lane >> 4) << 4;
  return *(const v16bf*)(src + (size_t)n * ld + k0 + kg);
}

// Row reductions across a 16-lane half (C/D rows live in lanes [g*16, g*16+15]).
static __device__ __forceinline__ float rowmax16(float x) {
#pragma unroll
  for (int m = 8; m >= 1; m >>= 1) x = fmaxf(x, __shfl_xor(x, m, 32));
  return x;
}
static __device__ __forceinline__ float rowsum16(float x) {
#pragma unroll
  for (int m = 8; m >= 1; m >>= 1) x += __shfl_xor(x, m, 32);
  return x;
}

// ---- Kernel 0a: one-time weight convert+transpose f32 KxN -> bf16 NxK ------
template <int K, int N>
__global__ void cvt_transpose_w(const float* __restrict__ in, bf16_t* __restrict__ out) {
  const size_t idx = (size_t)blockIdx.x * 256 + threadIdx.x;  // out index, k innermost
  const int    kk  = (int)(idx & (K - 1));
  const size_t r   = idx >> 10;           // K == 1024 always
  const int    n   = (int)(r & (N - 1));
  const size_t bat = r / N;
  out[idx] = (bf16_t)in[(bat * K + kk) * N + n];
}

// ---- Kernel 0b: one-time activation convert f32 -> bf16 (q,k,v) ------------
__global__ void cvt_x_bf16(const float* __restrict__ q, const float* __restrict__ k,
                           const float* __restrict__ v, bf16_t* __restrict__ xq,
                           bf16_t* __restrict__ xk, bf16_t* __restrict__ xv) {
  const size_t i8 = ((size_t)blockIdx.x * 256 + threadIdx.x) * 8;
  const float* src = (blockIdx.y == 0) ? q : (blockIdx.y == 1) ? k : v;
  bf16_t*      dst = (blockIdx.y == 0) ? xq : (blockIdx.y == 1) ? xk : xv;
  f32x8 x = *(const f32x8*)(src + i8);
  bf16x8 y;
#pragma unroll
  for (int i = 0; i < 8; ++i) y[i] = (bf16_t)x[i];
  *(bf16x8*)(dst + i8) = y;
}

// ---- Kernel 1: fused QKV projection (grid.z picks Q/K/V) -------------------
// Block = 4 waves, 32 output rows x 64 cols; each wave: 2 M-tiles x 1 N-tile,
// B-fragment shared across the two M-tiles.
__global__ void mha_proj_qkv(const bf16_t* __restrict__ xq, const bf16_t* __restrict__ xk,
                             const bf16_t* __restrict__ xv,
                             const bf16_t* __restrict__ WqT, const bf16_t* __restrict__ WkT,
                             const bf16_t* __restrict__ WvT,
                             const float* __restrict__ bq, const float* __restrict__ bk,
                             const float* __restrict__ bv,
                             bf16_t* __restrict__ Qo, bf16_t* __restrict__ Ko,
                             bf16_t* __restrict__ VTo) {
  const int lane  = threadIdx.x & 31;
  const int wv_   = threadIdx.x >> 5;   // 4 waves: N sub-tile
  const int m0    = blockIdx.x * 32;    // 2 row tiles over B*S
  const int h     = blockIdx.y;
  const int which = blockIdx.z;         // 0=Q 1=K 2=V

  const bf16_t* X  = (which == 0) ? xq : (which == 1) ? xk : xv;
  const bf16_t* Wt = ((which == 0) ? WqT : (which == 1) ? WkT : WvT)
                     + (size_t)h * DH_ * D_;                  // [Dh][D] bf16, K contiguous
  const float* bias = ((which == 0) ? bq : (which == 1) ? bk : bv) + h * DH_;
  const int n0 = wv_ * 16;

  v8f a0 = {}, a1 = {}, b0a = {}, b1a = {};
#pragma unroll 4
  for (int k0 = 0; k0 < D_; k0 += 64) {
    const v16bf w0 = load_Bfrag_rows_bf16(Wt, D_, n0, k0,      lane);
    const v16bf w1 = load_Bfrag_rows_bf16(Wt, D_, n0, k0 + 32, lane);
    a0  = wmma_bf16(load_Afrag_bf16(X, D_, m0,      k0,      lane), w0, a0);
    b0a = wmma_bf16(load_Afrag_bf16(X, D_, m0 + 16, k0,      lane), w0, b0a);
    a1  = wmma_bf16(load_Afrag_bf16(X, D_, m0,      k0 + 32, lane), w1, a1);
    b1a = wmma_bf16(load_Afrag_bf16(X, D_, m0 + 16, k0 + 32, lane), w1, b1a);
  }

  const int n    = n0 + (lane & 15);
  const int hf   = lane >> 4;
  const float bb = bias[n];
#pragma unroll
  for (int mt = 0; mt < 2; ++mt) {
#pragma unroll
    for (int j = 0; j < 8; ++j) {
      const int row = m0 + mt * 16 + j + hf * 8;  // global row in [0, B*S)
      const int bi  = row >> 11;                  // / S_
      const int s   = row & (S_ - 1);
      const float val = (mt ? (b0a[j] + b1a[j]) : (a0[j] + a1[j])) + bb;
      if (which == 2) {                           // V stored transposed [B,H,Dh,S]
        VTo[((size_t)(bi * H_ + h) * DH_ + n) * S_ + s] = (bf16_t)val;
      } else {                                    // Q,K stored [B,H,S,Dh]
        bf16_t* O = (which == 0) ? Qo : Ko;
        O[((size_t)(bi * H_ + h) * S_ + s) * DH_ + n] = (bf16_t)val;
      }
    }
  }
}

// ---- Kernel 2: flash attention -------------------------------------------
// 4 waves/block share one (b,h): K/V chunks staged in LDS once per block.
__global__ void mha_attn(const bf16_t* __restrict__ Q, const bf16_t* __restrict__ K,
                         const bf16_t* __restrict__ VT, bf16_t* __restrict__ Z) {
  __shared__ alignas(64) bf16_t ldsK[32 * 64];    // 32 key rows x 64 features (4KB)
  __shared__ alignas(64) bf16_t ldsV[64 * 32];    // 64 features x 32 key cols (4KB)
  __shared__ alignas(64) bf16_t plds[4 * 16 * 32];// per-wave P tile (4KB)

  const int t    = threadIdx.x;
  const int lane = t & 31;
  const int wv_  = t >> 5;
  const int wid  = blockIdx.x * 4 + wv_;
  const int qt   = wid & 127;                 // S_/16 q tiles; 4 consecutive per block
  const int h    = (wid >> 7) & (H_ - 1);     // same for all 4 waves in block
  const int b    = wid >> 11;
  const size_t bh = (size_t)(b * H_ + h);
  const bf16_t* Qb = Q  + bh * S_ * DH_;
  const bf16_t* Kb = K  + bh * S_ * DH_;
  const bf16_t* Vb = VT + bh * DH_ * S_;
  bf16_t* pw = plds + wv_ * (16 * 32);
  const int hf = lane >> 4;
  const int ln = lane & 15;

  const v16bf qa0 = load_Afrag_bf16(Qb, DH_, qt * 16, 0,  lane);
  const v16bf qa1 = load_Afrag_bf16(Qb, DH_, qt * 16, 32, lane);

  float rmax[8], rsum[8];
  v8f o0 = {}, o1 = {}, o2 = {}, o3 = {};
#pragma unroll
  for (int j = 0; j < 8; ++j) { rmax[j] = -INFINITY; rsum[j] = 0.f; }

  for (int s0 = 0; s0 < S_; s0 += 32) {
    __syncthreads();                              // previous chunk fully consumed
    // cooperative stage: K chunk (rows s0..s0+31, row-major, 4KB flat copy)
    {
      const bf16x8* srcv = (const bf16x8*)(Kb + (size_t)s0 * DH_);
      bf16x8*       dstv = (bf16x8*)ldsK;
      dstv[t]       = srcv[t];
      dstv[t + 128] = srcv[t + 128];
    }
    // cooperative stage: V^T chunk (64 rows x 32 cols); one 32B copy per thread
    {
      const int row = t >> 1, hh = (t & 1) * 16;
      *(v16bf*)(ldsV + row * 32 + hh) =
          *(const v16bf*)(Vb + (size_t)row * S_ + s0 + hh);
    }
    if (s0 + 32 < S_) {                           // prefetch next chunk into caches
      __builtin_prefetch((const char*)(Kb + (size_t)(s0 + 32) * DH_) + t * 32, 0, 1);
      __builtin_prefetch(Vb + (size_t)(t >> 1) * S_ + s0 + 32 + (t & 1) * 16, 0, 1);
    }
    __syncthreads();

    // scores: two 16x16 tiles, K-dim = 64 (2 WMMA each), B-frags from LDS
    v8f c0 = {}, c1 = {};
    {
      v16bf kb;
      kb = load_Bfrag_rows_bf16(ldsK, 64, 0,  0,  lane); c0 = wmma_bf16(qa0, kb, c0);
      kb = load_Bfrag_rows_bf16(ldsK, 64, 0,  32, lane); c0 = wmma_bf16(qa1, kb, c0);
      kb = load_Bfrag_rows_bf16(ldsK, 64, 16, 0,  lane); c1 = wmma_bf16(qa0, kb, c1);
      kb = load_Bfrag_rows_bf16(ldsK, 64, 16, 32, lane); c1 = wmma_bf16(qa1, kb, c1);
    }
    // online softmax over the 32-column block
#pragma unroll
    for (int j = 0; j < 8; ++j) {
      c0[j] *= 0.125f;                 // 1/sqrt(64)
      c1[j] *= 0.125f;
      float bm = rowmax16(fmaxf(c0[j], c1[j]));
      const float nm = fmaxf(rmax[j], bm);
      const float f  = __expf(rmax[j] - nm);
      rmax[j] = nm;
      rsum[j] *= f;
      o0[j] *= f; o1[j] *= f; o2[j] *= f; o3[j] *= f;
      const float p0 = __expf(c0[j] - nm);
      const float p1 = __expf(c1[j] - nm);
      pw[(j + hf * 8) * 32 + ln]      = (bf16_t)p0;   // stage P for A-frag layout
      pw[(j + hf * 8) * 32 + 16 + ln] = (bf16_t)p1;
      rsum[j] += rowsum16(p0 + p1);
    }
    // PV: A = P(16x32) from per-wave LDS, B = V^T chunk from shared LDS
    const v16bf pa = load_Afrag_bf16(pw, 32, 0, 0, lane);
    {
      v16bf vb;
      vb = load_Bfrag_rows_bf16(ldsV, 32, 0,  0, lane); o0 = wmma_bf16(pa, vb, o0);
      vb = load_Bfrag_rows_bf16(ldsV, 32, 16, 0, lane); o1 = wmma_bf16(pa, vb, o1);
      vb = load_Bfrag_rows_bf16(ldsV, 32, 32, 0, lane); o2 = wmma_bf16(pa, vb, o2);
      vb = load_Bfrag_rows_bf16(ldsV, 32, 48, 0, lane); o3 = wmma_bf16(pa, vb, o3);
    }
  }

  // normalize and store z as bf16 [B, S, H*Dh]
  bf16_t* Zb = Z + ((size_t)b * S_ + (size_t)qt * 16) * (H_ * DH_) + h * DH_;
#pragma unroll
  for (int j = 0; j < 8; ++j) {
    const float inv = 1.0f / rsum[j];
    const size_t r = (size_t)(j + hf * 8) * (H_ * DH_);
    Zb[r +  0 + ln] = (bf16_t)(o0[j] * inv);
    Zb[r + 16 + ln] = (bf16_t)(o1[j] * inv);
    Zb[r + 32 + ln] = (bf16_t)(o2[j] * inv);
    Zb[r + 48 + ln] = (bf16_t)(o3[j] * inv);
  }
}

// ---- Kernel 3: output projection  out = z @ W_O + b_O ----------------------
// 2 M-tiles per wave, B-fragment shared.
__global__ void mha_outproj(const bf16_t* __restrict__ Zw, const bf16_t* __restrict__ WoT,
                            const float* __restrict__ bO, float* __restrict__ out) {
  const int lane = threadIdx.x & 31;
  const int wv_  = threadIdx.x >> 5;
  const int m0   = blockIdx.x * 32;
  const int n0   = blockIdx.y * 64 + wv_ * 16;

  v8f a0 = {}, a1 = {}, b0a = {}, b1a = {};
#pragma unroll 4
  for (int k0 = 0; k0 < D_; k0 += 64) {
    const v16bf w0 = load_Bfrag_rows_bf16(WoT, D_, n0, k0,      lane);
    const v16bf w1 = load_Bfrag_rows_bf16(WoT, D_, n0, k0 + 32, lane);
    a0  = wmma_bf16(load_Afrag_bf16(Zw, D_, m0,      k0,      lane), w0, a0);
    b0a = wmma_bf16(load_Afrag_bf16(Zw, D_, m0 + 16, k0,      lane), w0, b0a);
    a1  = wmma_bf16(load_Afrag_bf16(Zw, D_, m0,      k0 + 32, lane), w1, a1);
    b1a = wmma_bf16(load_Afrag_bf16(Zw, D_, m0 + 16, k0 + 32, lane), w1, b1a);
  }
  const int col  = n0 + (lane & 15);
  const int hf   = lane >> 4;
  const float bb = bO[col];
#pragma unroll
  for (int j = 0; j < 8; ++j) {
    out[(size_t)(m0      + j + hf * 8) * D_ + col] = a0[j]  + a1[j]  + bb;
    out[(size_t)(m0 + 16 + j + hf * 8) * D_ + col] = b0a[j] + b1a[j] + bb;
  }
}

// ---- Host launcher ---------------------------------------------------------
extern "C" void kernel_launch(void* const* d_in, const int* in_sizes, int n_in,
                              void* d_out, int out_size, void* d_ws, size_t ws_size,
                              hipStream_t stream) {
  const float* query = (const float*)d_in[0];
  const float* key   = (const float*)d_in[1];
  const float* value = (const float*)d_in[2];
  const float* W_Q   = (const float*)d_in[3];
  const float* W_K   = (const float*)d_in[4];
  const float* W_V   = (const float*)d_in[5];
  const float* W_O   = (const float*)d_in[6];
  const float* b_Q   = (const float*)d_in[7];
  const float* b_K   = (const float*)d_in[8];
  const float* b_V   = (const float*)d_in[9];
  const float* b_O   = (const float*)d_in[10];
  float* out = (float*)d_out;

  bf16_t* qws  = (bf16_t*)d_ws;            // [B,H,S,Dh]     8MB
  bf16_t* kws  = qws  + PROJ_ELEMS;        // [B,H,S,Dh]     8MB
  bf16_t* vtws = kws  + PROJ_ELEMS;        // [B,H,Dh,S]     8MB
  bf16_t* zws  = vtws + PROJ_ELEMS;        // [B,S,H*Dh]     8MB
  bf16_t* wqt  = zws  + PROJ_ELEMS;        // [H,Dh,D] bf16  2MB
  bf16_t* wkt  = wqt  + WQKV_ELEMS;
  bf16_t* wvt  = wkt  + WQKV_ELEMS;
  bf16_t* wot  = wvt  + WQKV_ELEMS;        // [D,H*Dh] bf16  2MB
  bf16_t* xq   = wot  + X_ELEMS / 4;       // wot is D_*D_ = 1M elems; xq after it
  bf16_t* xk   = xq   + X_ELEMS;           // [B*S, D] bf16  8MB each
  bf16_t* xv   = xk   + X_ELEMS;

  // 0) one-time converts: weights (transposed, K-contiguous) + activations
  cvt_transpose_w<D_, DH_><<<dim3(WQKV_ELEMS / 256), 256, 0, stream>>>(W_Q, wqt);
  cvt_transpose_w<D_, DH_><<<dim3(WQKV_ELEMS / 256), 256, 0, stream>>>(W_K, wkt);
  cvt_transpose_w<D_, DH_><<<dim3(WQKV_ELEMS / 256), 256, 0, stream>>>(W_V, wvt);
  cvt_transpose_w<D_, D_ ><<<dim3((size_t)D_ * D_ / 256), 256, 0, stream>>>(W_O, wot);
  cvt_x_bf16<<<dim3(X_ELEMS / (256 * 8), 3), 256, 0, stream>>>(query, key, value,
                                                               xq, xk, xv);

  // 1) QKV projection: 128 row-blocks x 16 heads x {Q,K,V}
  mha_proj_qkv<<<dim3(BS_ / 32, H_, 3), 128, 0, stream>>>(
      xq, xk, xv, wqt, wkt, wvt, b_Q, b_K, b_V, qws, kws, vtws);

  // 2) attention: B*H*(S/16) = 4096 waves, 4 waves/block sharing K/V via LDS
  mha_attn<<<dim3((BS_ / 16) * H_ / 4), 128, 0, stream>>>(qws, kws, vtws, zws);

  // 3) output projection: 128 row-blocks x 16 column-groups
  mha_outproj<<<dim3(BS_ / 32, D_ / 64), 128, 0, stream>>>(zws, wot, b_O, out);
}
